// SoluModel_7172595385032
// MI455X (gfx1250) — matrix operand order repaired and checked by reading, not compiled, not run on hardware
//
#include <hip/hip_runtime.h>
#include <hip/hip_bf16.h>
#include <math.h>

#define BB   8
#define LL   1024
#define CC   768
#define HH   4
#define DD   192          // CC / HH
#define CMID 1536
#define CLSZ 2304
#define FC1N 576

typedef __bf16 bf16;
typedef __attribute__((ext_vector_type(16))) __bf16 v16bf;
typedef __attribute__((ext_vector_type(8)))  __bf16 v8bf;
typedef __attribute__((ext_vector_type(8)))  float  v8f;

__device__ __forceinline__ unsigned short f32_to_bf16(float f) {
    unsigned u = __float_as_uint(f);
    u += 0x7FFFu + ((u >> 16) & 1u);      // round-to-nearest-even
    return (unsigned short)(u >> 16);
}
__device__ __forceinline__ float bf16u_to_f32(unsigned short u) {
    return __uint_as_float(((unsigned)u) << 16);
}

// ---------------------------------------------------------------------------
// Elementwise converts / packing
// ---------------------------------------------------------------------------
__global__ void cvt_f32_bf16(const float* __restrict__ x, unsigned short* __restrict__ y, long long n) {
    long long i = (long long)blockIdx.x * 256 + threadIdx.x;
    if (i < n) y[i] = f32_to_bf16(x[i]);
}

// w [K, N] f32 -> wt [N, K] bf16   (for QKV projection weights)
__global__ void cvt_transpose_kn(const float* __restrict__ w, unsigned short* __restrict__ wt,
                                 int K, int N) {
    long long i = (long long)blockIdx.x * 256 + threadIdx.x;
    if (i >= (long long)K * N) return;
    int k = (int)(i % K);
    int n = (int)(i / K);
    wt[i] = f32_to_bf16(w[(long long)k * N + n]);
}

// w [Cout, Cin, 3] f32 -> wt [Cout, 3*Cin] bf16  (col index = k*Cin + ci)  => B in [N,K]
__global__ void convw_transpose(const float* __restrict__ w, unsigned short* __restrict__ wt,
                                int Cin, int Cout) {
    long long i = (long long)blockIdx.x * 256 + threadIdx.x;
    long long n = (long long)3 * Cin * Cout;
    if (i >= n) return;
    int ci = (int)(i % Cin);
    int k  = (int)((i / Cin) % 3);
    int co = (int)(i / (3LL * Cin));
    wt[i] = f32_to_bf16(w[((long long)co * Cin + ci) * 3 + k]);
}

// x [B, L, C] f32 -> xt [B, C, L] bf16 (transpose-convert; used for V^T)
__global__ void transpose_cvt_blc(const float* __restrict__ x, unsigned short* __restrict__ xt) {
    long long i = (long long)blockIdx.x * 256 + threadIdx.x;
    long long n = (long long)BB * CC * LL;
    if (i >= n) return;
    int l = (int)(i % LL);
    int c = (int)((i / LL) % CC);
    int b = (int)(i / ((long long)CC * LL));
    xt[i] = f32_to_bf16(x[((long long)b * LL + l) * CC + c]);
}

// x [B, L, Cin] f32 -> col [B*L, 3*Cin] bf16 with zero padding at sequence ends
__global__ void im2col3(const float* __restrict__ x, unsigned short* __restrict__ col, int Cin) {
    long long i = (long long)blockIdx.x * 256 + threadIdx.x;
    long long n = (long long)BB * LL * 3 * Cin;
    if (i >= n) return;
    int ci = (int)(i % Cin);
    int k  = (int)((i / Cin) % 3);
    long long row = i / (3LL * Cin);
    int l = (int)(row % LL);
    int b = (int)(row / LL);
    int ls = l + k - 1;
    float v = (ls >= 0 && ls < LL) ? x[((long long)b * LL + ls) * Cin + ci] : 0.f;
    col[i] = f32_to_bf16(v);
}

// ---------------------------------------------------------------------------
// Batched WMMA GEMM (NT only): C[M,N] = alpha * A[M,K](bf16) x B[N,K](bf16)^T + bias
// Per-z operand offsets: off = (z/zdiv)*s0 + (z%zdiv)*s1   (in elements)
// Block = 128 threads (4 waves); block tile 128(M) x 64(N); wave tile 32 x 64.
// Fragment layouts used here (A/B k order is this code's choice, not the ISA table):
//   A 16x32: lane half selects K 0-15/16-31 contiguous, row = lane&15
//   B 32x16: lane half selects K 0-15/16-31 contiguous, col = lane&15
//   C/D    : VGPR r -> row = r + 8*(lane>=16), col = lane&15
// ---------------------------------------------------------------------------
__global__ __launch_bounds__(128) void wmma_gemm(
    const unsigned short* __restrict__ Abuf, const unsigned short* __restrict__ Bbuf,
    const float* __restrict__ bias, float* __restrict__ Cbuf,
    int M, int N, int K, int lda, int ldb, int ldc, float alpha,
    int zdiv, long long sA0, long long sA1, long long sB0, long long sB1,
    long long sC0, long long sC1)
{
    const int z = blockIdx.z;
    const bf16* A  = (const bf16*)Abuf + (long long)(z / zdiv) * sA0 + (long long)(z % zdiv) * sA1;
    const bf16* Bm = (const bf16*)Bbuf + (long long)(z / zdiv) * sB0 + (long long)(z % zdiv) * sB1;
    float*      Cp = Cbuf             + (long long)(z / zdiv) * sC0 + (long long)(z % zdiv) * sC1;

    const int lane = threadIdx.x & 31;
    const int wave = threadIdx.x >> 5;
    const int m0   = blockIdx.y * 128 + wave * 32;
    const int n0   = blockIdx.x * 64;
    const int arow = lane & 15;      // A row / B column within 16-tile
    const int half = lane >> 4;      // lane group selector

    v8f acc[2][4] = {{v8f{}, v8f{}, v8f{}, v8f{}}, {v8f{}, v8f{}, v8f{}, v8f{}}};

    for (int kc = 0; kc < K; kc += 32) {
        // ---- two A fragments (rows m0.., m0+16..)
        v16bf afrag[2];
#pragma unroll
        for (int mi = 0; mi < 2; mi++) {
            const bf16* ap = A + (long long)(m0 + 16 * mi + arow) * lda + kc + half * 16;
            __builtin_prefetch((const void*)(ap + 512), 0, 1);   // global_prefetch_b8
            v8bf alo = *(const v8bf*)ap;
            v8bf ahi = *(const v8bf*)(ap + 8);
            v16bf a;
#pragma unroll
            for (int i = 0; i < 8; i++) { a[i] = alo[i]; a[i + 8] = ahi[i]; }
            afrag[mi] = a;
        }
        // ---- four B fragments: contiguous 32B per lane (B stored [N,K])
        v16bf bfrag[4];
#pragma unroll
        for (int t = 0; t < 4; t++) {
            const bf16* bp = Bm + (long long)(n0 + 16 * t + arow) * ldb + kc + half * 16;
            bfrag[t] = *(const v16bf*)bp;
        }
#pragma unroll
        for (int mi = 0; mi < 2; mi++)
#pragma unroll
            for (int t = 0; t < 4; t++)
                acc[mi][t] = __builtin_amdgcn_wmma_f32_16x16x32_bf16(
                    false, afrag[mi], false, bfrag[t], (short)0, acc[mi][t], false, false);
    }

    // ---- store
#pragma unroll
    for (int t = 0; t < 4; t++) {
        int n = n0 + 16 * t + arow;
        float bv = bias ? bias[n] : 0.f;
#pragma unroll
        for (int mi = 0; mi < 2; mi++) {
#pragma unroll
            for (int r = 0; r < 8; r++) {
                Cp[(long long)(m0 + 16 * mi + half * 8 + r) * ldc + n] = alpha * acc[mi][t][r] + bv;
            }
        }
    }
}

// Scalar GEMM for tiny N (the Cout=2 conv head); B stored [N,K] -> both operands contiguous
__global__ void gemm_smallN(const unsigned short* __restrict__ A, const unsigned short* __restrict__ BT,
                            const float* __restrict__ bias, float* __restrict__ C,
                            int M, int N, int K) {
    long long t = (long long)blockIdx.x * 256 + threadIdx.x;
    if (t >= (long long)M * N) return;
    int n = (int)(t % N);
    long long m = t / N;
    float acc = bias ? bias[n] : 0.f;
    const unsigned short* a = A + m * K;
    const unsigned short* b = BT + (long long)n * K;
    for (int k = 0; k < K; k++)
        acc += bf16u_to_f32(a[k]) * bf16u_to_f32(b[k]);
    C[m * N + n] = acc;
}

// ---------------------------------------------------------------------------
// LayerNorm over (C,L) per batch + LeakyReLU
// ---------------------------------------------------------------------------
__global__ void ln_reduce(const float* __restrict__ x, long long perBatch, float* __restrict__ stats) {
    __shared__ float ss[256], s2[256];
    int b = blockIdx.x;
    const float* p = x + (long long)b * perBatch;
    float s = 0.f, q = 0.f;
    for (long long i = threadIdx.x; i < perBatch; i += 256) { float v = p[i]; s += v; q += v * v; }
    ss[threadIdx.x] = s; s2[threadIdx.x] = q; __syncthreads();
    for (int o = 128; o > 0; o >>= 1) {
        if ((int)threadIdx.x < o) { ss[threadIdx.x] += ss[threadIdx.x + o]; s2[threadIdx.x] += s2[threadIdx.x + o]; }
        __syncthreads();
    }
    if (threadIdx.x == 0) {
        float mu = ss[0] / (float)perBatch;
        float var = s2[0] / (float)perBatch - mu * mu;
        stats[2 * b] = mu;
        stats[2 * b + 1] = rsqrtf(var + 1e-5f);
    }
}

__global__ void ln_apply_lrelu(const float* __restrict__ y, const float* __restrict__ stats,
                               const float* __restrict__ g, const float* __restrict__ be,
                               float* __restrict__ out, int Cc) {
    long long i = (long long)blockIdx.x * 256 + threadIdx.x;
    long long n = (long long)BB * LL * Cc;
    if (i >= n) return;
    int c = (int)(i % Cc);
    int l = (int)((i / Cc) % LL);
    int b = (int)(i / ((long long)Cc * LL));
    float v = (y[i] - stats[2 * b]) * stats[2 * b + 1] * g[(long long)c * LL + l] + be[(long long)c * LL + l];
    out[i] = v >= 0.f ? v : 0.01f * v;
}

// ---------------------------------------------------------------------------
// Softmax over rows of length n (scores -> bf16 probs)
// ---------------------------------------------------------------------------
__global__ void softmax_row(const float* __restrict__ S, unsigned short* __restrict__ P, int n) {
    __shared__ float red[256];
    long long row = blockIdx.x;
    const float* s = S + row * n;
    int t = threadIdx.x;
    float m = -3.4e38f;
    for (int i = t; i < n; i += 256) m = fmaxf(m, s[i]);
    red[t] = m; __syncthreads();
    for (int o = 128; o > 0; o >>= 1) { if (t < o) red[t] = fmaxf(red[t], red[t + o]); __syncthreads(); }
    m = red[0]; __syncthreads();
    float sum = 0.f;
    for (int i = t; i < n; i += 256) sum += __expf(s[i] - m);
    red[t] = sum; __syncthreads();
    for (int o = 128; o > 0; o >>= 1) { if (t < o) red[t] += red[t + o]; __syncthreads(); }
    float inv = 1.f / red[0];
    unsigned short* p = P + row * n;
    for (int i = t; i < n; i += 256) p[i] = f32_to_bf16(__expf(s[i] - m) * inv);
}

// softmax along L for s/w channels of fx [B, L, 2]
__global__ void softmax_sw(const float* __restrict__ fx, float* __restrict__ ssoft, float* __restrict__ wsoft) {
    __shared__ float red[256];
    int b = blockIdx.x, ch = blockIdx.y, t = threadIdx.x;
    const float* base = fx + (long long)b * LL * 2 + ch;
    float m = -3.4e38f;
    for (int i = t; i < LL; i += 256) m = fmaxf(m, base[2 * i]);
    red[t] = m; __syncthreads();
    for (int o = 128; o > 0; o >>= 1) { if (t < o) red[t] = fmaxf(red[t], red[t + o]); __syncthreads(); }
    m = red[0]; __syncthreads();
    float sum = 0.f;
    for (int i = t; i < LL; i += 256) sum += __expf(base[2 * i] - m);
    red[t] = sum; __syncthreads();
    for (int o = 128; o > 0; o >>= 1) { if (t < o) red[t] += red[t + o]; __syncthreads(); }
    float inv = 1.f / red[0];
    float* out = (ch ? wsoft : ssoft) + (long long)b * LL;
    for (int i = t; i < LL; i += 256) out[i] = __expf(base[2 * i] - m) * inv;
}

// G[b,i,j] = exp(-(i-j)^2 / (1e-5 + 2*std^2)); row-max normalization cancels analytically
__global__ void gauss_kernel(const float* __restrict__ ssoft, unsigned short* __restrict__ G) {
    long long i = (long long)blockIdx.x * 256 + threadIdx.x;
    long long n = (long long)BB * LL * LL;
    if (i >= n) return;
    int j = (int)(i % LL);
    long long r = i / LL;
    int ci = (int)(r % LL);
    int b  = (int)(r / LL);
    float std = 0.1f * (float)LL * ssoft[(long long)b * LL + ci];
    float inv = 1.f / (1e-5f + 2.f * std * std);
    float d = (float)(ci - j);
    G[i] = f32_to_bf16(__expf(-d * d * inv));
}

// wfT[b, d, j] = w[b,j] * feats[b,j,d]  (transposed for NT GEMM B operand)
__global__ void wf_kernel(const float* __restrict__ feats, const float* __restrict__ wsoft,
                          unsigned short* __restrict__ wfT) {
    long long i = (long long)blockIdx.x * 256 + threadIdx.x;
    long long n = (long long)BB * CC * LL;
    if (i >= n) return;
    int j = (int)(i % LL);
    int d = (int)((i / LL) % CC);
    int b = (int)(i / ((long long)CC * LL));
    wfT[i] = f32_to_bf16(wsoft[(long long)b * LL + j] * feats[((long long)b * LL + j) * CC + d]);
}

// max over L: cat[b, offset + c] = max_l (x[b,l,c] (+ add[b,l,c]))
__global__ void maxpool_l(const float* __restrict__ x, const float* __restrict__ addp,
                          float* __restrict__ cat, int Cc, int offset) {
    int t = blockIdx.x * 256 + threadIdx.x;
    if (t >= BB * Cc) return;
    int b = t / Cc, c = t % Cc;
    float m = -3.4e38f;
    for (int l = 0; l < LL; l++) {
        long long idx = ((long long)b * LL + l) * Cc + c;
        float v = x[idx];
        if (addp) v += addp[idx];
        m = fmaxf(m, v);
    }
    cat[(long long)b * CLSZ + offset + c] = m;
}

// BatchNorm1d over batch dim (training stats), write normalized "out" (2nd output)
__global__ void batchnorm_head(const float* __restrict__ cat, const float* __restrict__ g,
                               const float* __restrict__ be, float* __restrict__ outp) {
    int c = blockIdx.x * 256 + threadIdx.x;
    if (c >= CLSZ) return;
    float mu = 0.f;
    for (int b = 0; b < BB; b++) mu += cat[(long long)b * CLSZ + c];
    mu *= (1.f / BB);
    float var = 0.f;
    for (int b = 0; b < BB; b++) { float d = cat[(long long)b * CLSZ + c] - mu; var += d * d; }
    var *= (1.f / BB);
    float rs = rsqrtf(var + 1e-5f);
    for (int b = 0; b < BB; b++)
        outp[(long long)b * CLSZ + c] = (cat[(long long)b * CLSZ + c] - mu) * rs * g[c] + be[c];
}

__global__ void fc1_kernel(const float* __restrict__ x, const float* __restrict__ w,
                           const float* __restrict__ b, float* __restrict__ h) {
    int t = blockIdx.x * 256 + threadIdx.x;
    if (t >= BB * FC1N) return;
    int bb = t / FC1N, j = t % FC1N;
    float acc = b[j];
    for (int c = 0; c < CLSZ; c++) acc += x[(long long)bb * CLSZ + c] * w[(long long)c * FC1N + j];
    h[t] = acc > 0.f ? acc : 0.f;
}

__global__ void fc2_kernel(const float* __restrict__ h, const float* __restrict__ w,
                           const float* __restrict__ b, float* __restrict__ cls) {
    int t = blockIdx.x * 256 + threadIdx.x;
    if (t >= BB) return;
    float acc = b[0];
    for (int j = 0; j < FC1N; j++) acc += h[(long long)t * FC1N + j] * w[j];
    cls[t] = acc;
}

// ---------------------------------------------------------------------------
// Host orchestration
// ---------------------------------------------------------------------------
extern "C" void kernel_launch(void* const* d_in, const int* in_sizes, int n_in,
                              void* d_out, int out_size, void* d_ws, size_t ws_size,
                              hipStream_t stream) {
    (void)in_sizes; (void)n_in; (void)out_size; (void)ws_size;
    const float* feats = (const float*)d_in[0];
    const float* cpP[12]; for (int i = 0; i < 12; i++) cpP[i] = (const float*)d_in[1 + i];
    const float* cvP[12]; for (int i = 0; i < 12; i++) cvP[i] = (const float*)d_in[13 + i];
    const float* wq = (const float*)d_in[25]; const float* bq = (const float*)d_in[26];
    const float* wk = (const float*)d_in[27]; const float* bk = (const float*)d_in[28];
    const float* wv = (const float*)d_in[29]; const float* bv = (const float*)d_in[30];
    const float* bn_g = (const float*)d_in[31]; const float* bn_b = (const float*)d_in[32];
    const float* fc1_w = (const float*)d_in[33]; const float* fc1_b = (const float*)d_in[34];
    const float* fc2_w = (const float*)d_in[35]; const float* fc2_b = (const float*)d_in[36];
    float* outp = (float*)d_out;

    // bump allocator over workspace
    char* ws = (char*)d_ws; size_t off = 0;
    auto alloc = [&](size_t bytes) -> void* {
        void* p = ws + off; off += (bytes + 255) & ~(size_t)255; return p;
    };
    const long long ML = (long long)BB * LL;                   // 8192 rows
    unsigned short* xbf   = (unsigned short*)alloc(ML * CC * 2);
    unsigned short* wqT   = (unsigned short*)alloc((size_t)CC * CC * 2);  // [N,K]
    unsigned short* wkT   = (unsigned short*)alloc((size_t)CC * CC * 2);
    unsigned short* wvT   = (unsigned short*)alloc((size_t)CC * CC * 2);
    unsigned short* cpw1t = (unsigned short*)alloc((size_t)CMID * 3 * CC * 2);    // [Cout, 3Cin]
    unsigned short* cpw2t = (unsigned short*)alloc((size_t)CMID * 3 * CMID * 2);
    unsigned short* cpw3t = (unsigned short*)alloc((size_t)2 * 3 * CMID * 2);
    unsigned short* cvw1t = (unsigned short*)alloc((size_t)CMID * 3 * CC * 2);
    unsigned short* cvw2t = (unsigned short*)alloc((size_t)CMID * 3 * CMID * 2);
    unsigned short* cvw3t = (unsigned short*)alloc((size_t)CC * 3 * CMID * 2);
    float* qf = (float*)alloc(ML * CC * 4);
    float* kf = (float*)alloc(ML * CC * 4);
    float* vf = (float*)alloc(ML * CC * 4);
    unsigned short* qb = (unsigned short*)alloc(ML * CC * 2);
    unsigned short* kb = (unsigned short*)alloc(ML * CC * 2);
    unsigned short* vT = (unsigned short*)alloc(ML * CC * 2);              // [B, C, L]
    float* scores = (float*)alloc((size_t)BB * HH * LL * LL * 4);
    unsigned short* probs = (unsigned short*)alloc((size_t)BB * HH * LL * LL * 2);
    float* out_sa = (float*)alloc(ML * CC * 4);
    float* ybuf  = (float*)alloc(ML * CMID * 4);
    float* ynorm = (float*)alloc(ML * CMID * 4);
    unsigned short* col = (unsigned short*)alloc(ML * 3 * CMID * 2);
    float* stats = (float*)alloc(2 * BB * 4);
    float* ssoft = (float*)alloc((size_t)BB * LL * 4);
    float* wsoft = (float*)alloc((size_t)BB * LL * 4);
    unsigned short* wfT = (unsigned short*)alloc(ML * CC * 2);             // [B, C, L]
    unsigned short* Gb  = (unsigned short*)alloc((size_t)BB * LL * LL * 2);
    float* out_cp = (float*)alloc(ML * CC * 4);
    float* catb = (float*)alloc((size_t)BB * CLSZ * 4);
    float* hbuf = (float*)alloc((size_t)BB * FC1N * 4);

    auto g1d = [](long long n) { return dim3((unsigned)((n + 255) / 256)); };

    // ---------------- conversions & weight packing ----------------
    cvt_f32_bf16<<<g1d(ML * CC), 256, 0, stream>>>(feats, xbf, ML * CC);
    cvt_transpose_kn<<<g1d((long long)CC * CC), 256, 0, stream>>>(wq, wqT, CC, CC);
    cvt_transpose_kn<<<g1d((long long)CC * CC), 256, 0, stream>>>(wk, wkT, CC, CC);
    cvt_transpose_kn<<<g1d((long long)CC * CC), 256, 0, stream>>>(wv, wvT, CC, CC);
    convw_transpose<<<g1d((long long)3 * CC * CMID),   256, 0, stream>>>(cpP[0], cpw1t, CC,   CMID);
    convw_transpose<<<g1d((long long)3 * CMID * CMID), 256, 0, stream>>>(cpP[4], cpw2t, CMID, CMID);
    convw_transpose<<<g1d((long long)3 * CMID * 2),    256, 0, stream>>>(cpP[8], cpw3t, CMID, 2);
    convw_transpose<<<g1d((long long)3 * CC * CMID),   256, 0, stream>>>(cvP[0], cvw1t, CC,   CMID);
    convw_transpose<<<g1d((long long)3 * CMID * CMID), 256, 0, stream>>>(cvP[4], cvw2t, CMID, CMID);
    convw_transpose<<<g1d((long long)3 * CMID * CC),   256, 0, stream>>>(cvP[8], cvw3t, CMID, CC);

    // ---------------- self-attention branch ----------------
    {
        dim3 g(CC / 64, (unsigned)(ML / 128), 1);
        wmma_gemm<<<g, 128, 0, stream>>>(xbf, wqT, bq, qf, (int)ML, CC, CC, CC, CC, CC, 1.f,
                                         1, 0, 0, 0, 0, 0, 0);
        wmma_gemm<<<g, 128, 0, stream>>>(xbf, wkT, bk, kf, (int)ML, CC, CC, CC, CC, CC, 1.f,
                                         1, 0, 0, 0, 0, 0, 0);
        wmma_gemm<<<g, 128, 0, stream>>>(xbf, wvT, bv, vf, (int)ML, CC, CC, CC, CC, CC, 1.f,
                                         1, 0, 0, 0, 0, 0, 0);
    }
    cvt_f32_bf16<<<g1d(ML * CC), 256, 0, stream>>>(qf, qb, ML * CC);
    cvt_f32_bf16<<<g1d(ML * CC), 256, 0, stream>>>(kf, kb, ML * CC);
    transpose_cvt_blc<<<g1d(ML * CC), 256, 0, stream>>>(vf, vT);
    {   // scores = Q Kt / sqrt(D): batched NT GEMM over z = b*H + h (K rows are Kt columns)
        float alpha = 1.f / sqrtf((float)DD);
        dim3 g(LL / 64, LL / 128, BB * HH);
        wmma_gemm<<<g, 128, 0, stream>>>(qb, kb, nullptr, scores, LL, LL, DD, CC, CC, LL, alpha,
                                         HH, (long long)LL * CC, DD,
                                             (long long)LL * CC, DD,
                                             (long long)HH * LL * LL, (long long)LL * LL);
    }
    softmax_row<<<BB * HH * LL, 256, 0, stream>>>(scores, probs, LL);
    {   // out = P V : NT GEMM with B = V^T [D, L] slices of vT [B, C, L]
        dim3 g(DD / 64, LL / 128, BB * HH);
        wmma_gemm<<<g, 128, 0, stream>>>(probs, vT, nullptr, out_sa, LL, DD, LL, LL, LL, CC, 1.f,
                                         HH, (long long)HH * LL * LL, (long long)LL * LL,
                                             (long long)CC * LL, (long long)DD * LL,
                                             (long long)LL * CC, DD);
    }
    maxpool_l<<<g1d(BB * CC), 256, 0, stream>>>(out_sa, nullptr, catb, CC, 0);

    // ---------------- conv block runner ----------------
    auto conv_block = [&](const float* P[12], unsigned short* w1t, unsigned short* w2t,
                          unsigned short* w3t, int Cout3, bool small3) {
        // stage 1: Cin = CC
        im2col3<<<g1d(ML * 3 * CC), 256, 0, stream>>>(feats, col, CC);
        wmma_gemm<<<dim3(CMID / 64, (unsigned)(ML / 128), 1), 128, 0, stream>>>(
            col, w1t, P[1], ybuf, (int)ML, CMID, 3 * CC, 3 * CC, 3 * CC, CMID, 1.f,
            1, 0, 0, 0, 0, 0, 0);
        ln_reduce<<<BB, 256, 0, stream>>>(ybuf, (long long)LL * CMID, stats);
        ln_apply_lrelu<<<g1d(ML * CMID), 256, 0, stream>>>(ybuf, stats, P[2], P[3], ynorm, CMID);
        // stage 2: Cin = CMID
        im2col3<<<g1d(ML * 3 * CMID), 256, 0, stream>>>(ynorm, col, CMID);
        wmma_gemm<<<dim3(CMID / 64, (unsigned)(ML / 128), 1), 128, 0, stream>>>(
            col, w2t, P[5], ybuf, (int)ML, CMID, 3 * CMID, 3 * CMID, 3 * CMID, CMID, 1.f,
            1, 0, 0, 0, 0, 0, 0);
        ln_reduce<<<BB, 256, 0, stream>>>(ybuf, (long long)LL * CMID, stats);
        ln_apply_lrelu<<<g1d(ML * CMID), 256, 0, stream>>>(ybuf, stats, P[6], P[7], ynorm, CMID);
        // stage 3
        im2col3<<<g1d(ML * 3 * CMID), 256, 0, stream>>>(ynorm, col, CMID);
        if (small3) {
            gemm_smallN<<<g1d(ML * Cout3), 256, 0, stream>>>(col, w3t, P[9], ybuf,
                                                             (int)ML, Cout3, 3 * CMID);
        } else {
            wmma_gemm<<<dim3(Cout3 / 64, (unsigned)(ML / 128), 1), 128, 0, stream>>>(
                col, w3t, P[9], ybuf, (int)ML, Cout3, 3 * CMID, 3 * CMID, 3 * CMID, Cout3, 1.f,
                1, 0, 0, 0, 0, 0, 0);
        }
        ln_reduce<<<BB, 256, 0, stream>>>(ybuf, (long long)LL * Cout3, stats);
        ln_apply_lrelu<<<g1d(ML * Cout3), 256, 0, stream>>>(ybuf, stats, P[10], P[11], ynorm, Cout3);
    };

    // ---------------- ContextPooling branch ----------------
    conv_block(cpP, cpw1t, cpw2t, cpw3t, 2, true);          // ynorm[:, :, 0:2] = fx
    softmax_sw<<<dim3(BB, 2), 256, 0, stream>>>(ynorm, ssoft, wsoft);
    wf_kernel<<<g1d(ML * CC), 256, 0, stream>>>(feats, wsoft, wfT);
    gauss_kernel<<<g1d((long long)BB * LL * LL), 256, 0, stream>>>(ssoft, Gb);
    {   // out_cp = G @ (w*feats): NT GEMM with B = wfT [C, L] per batch
        dim3 g(CC / 64, LL / 128, BB);
        wmma_gemm<<<g, 128, 0, stream>>>(Gb, wfT, nullptr, out_cp, LL, CC, LL, LL, LL, CC, 1.f,
                                         1, (long long)LL * LL, 0,
                                            (long long)CC * LL, 0,
                                            (long long)LL * CC, 0);
    }
    maxpool_l<<<g1d(BB * CC), 256, 0, stream>>>(out_cp, nullptr, catb, CC, 2 * CC);

    // ---------------- conv branch (+ residual with feats) ----------------
    conv_block(cvP, cvw1t, cvw2t, cvw3t, CC, false);        // ynorm = conv output [B,L,CC]
    maxpool_l<<<g1d(BB * CC), 256, 0, stream>>>(ynorm, feats, catb, CC, CC);

    // ---------------- head: BatchNorm (output #2) + classifier (output #1) ----------------
    batchnorm_head<<<g1d(CLSZ), 256, 0, stream>>>(catb, bn_g, bn_b, outp + BB);
    fc1_kernel<<<g1d(BB * FC1N), 256, 0, stream>>>(outp + BB, fc1_w, fc1_b, hbuf);
    fc2_kernel<<<1, 256, 0, stream>>>(hbuf, fc2_w, fc2_b, outp);
}
